// GPT_16277926052595
// MI455X (gfx1250) — matrix operand order repaired and checked by reading, not compiled, not run on hardware
//
#include <hip/hip_runtime.h>
#include <hip/hip_bf16.h>
#include <cmath>

// ---------------- problem constants (from reference) ----------------
#define Lc  6
#define Hc  12
#define Ec  768
#define HSc 64
#define Tc  1024
#define Bc  2
#define Vc  32000
#define BTc (Bc * Tc)   // 2048

typedef __bf16 bf16;
typedef bf16  v16bf __attribute__((ext_vector_type(16)));
typedef bf16  v8bf  __attribute__((ext_vector_type(8)));
typedef bf16  v4bf  __attribute__((ext_vector_type(4)));
typedef float v8f   __attribute__((ext_vector_type(8)));

#define KT   64   // K-step per staging round (2 WMMA sub-steps)
#define LDSP 72   // padded k-stride (bf16): 144B = 36-bank stride -> conflict-free frags

// =====================================================================
// Tiled bf16 WMMA GEMM:  C[M,N] = act(A[M,K] @ W[K,N] + bias[N])
// Block = 256 threads (8 wave32), tile 128x128, K-step 64.
// Waves 4x2; each wave computes 32x64 = 2x4 WMMA 16x16 tiles.
// fp32 -> bf16 conversion happens while staging to LDS.
// Software-pipelined: next tile's global loads issue before current
// tile's WMMA compute, hiding load latency under matrix ops.
// =====================================================================
__global__ __launch_bounds__(256)
void gemm_bf16_wmma(const float* __restrict__ A, const float* __restrict__ W,
                    const float* __restrict__ bias, float* __restrict__ C,
                    int M, int N, int K, int relu) {
    __shared__ bf16 As[128 * LDSP];   // [row][k]
    __shared__ bf16 Bs[128 * LDSP];   // [n][k]  (W tile transposed)

    const int tid    = threadIdx.x;
    const int lane   = tid & 31;
    const int wid    = tid >> 5;
    const int wave_m = wid >> 1;        // 0..3
    const int wave_n = wid & 1;         // 0..1
    const int half   = lane >> 4;
    const int l16    = lane & 15;

    const int base_m = blockIdx.y * 128;
    const int base_n = blockIdx.x * 128;

    // Per-thread staging work: 8 chunks of A (float4 each) + 8 chunks of B.
    // A chunk c: ci = tid + 256*c ; row = ci/16 ; k = (ci%16)*4   (coalesced b128)
    // B chunk c: ci = tid + 256*c ; n = ci%128 ; k = (ci/128)*4   (4 coalesced b32)
    float4 aReg[8];
    float  bReg[8][4];

    auto loadTile = [&](int k0) {
#pragma unroll
        for (int c = 0; c < 8; ++c) {
            int ci = tid + 256 * c;
            int r  = ci >> 4;
            int kq = (ci & 15) << 2;
            aReg[c] = *(const float4*)&A[(size_t)(base_m + r) * K + (k0 + kq)];
        }
#pragma unroll
        for (int c = 0; c < 8; ++c) {
            int ci = tid + 256 * c;
            int n  = ci & 127;
            int kk = (ci >> 7) << 2;
#pragma unroll
            for (int j = 0; j < 4; ++j)
                bReg[c][j] = W[(size_t)(k0 + kk + j) * N + (base_n + n)];
        }
    };

    auto storeTile = [&]() {
#pragma unroll
        for (int c = 0; c < 8; ++c) {
            int ci = tid + 256 * c;
            int r  = ci >> 4;
            int kq = (ci & 15) << 2;
            v4bf p;
            p[0] = (bf16)aReg[c].x; p[1] = (bf16)aReg[c].y;
            p[2] = (bf16)aReg[c].z; p[3] = (bf16)aReg[c].w;
            *(v4bf*)&As[r * LDSP + kq] = p;
        }
#pragma unroll
        for (int c = 0; c < 8; ++c) {
            int ci = tid + 256 * c;
            int n  = ci & 127;
            int kk = (ci >> 7) << 2;
            v4bf p;
            p[0] = (bf16)bReg[c][0]; p[1] = (bf16)bReg[c][1];
            p[2] = (bf16)bReg[c][2]; p[3] = (bf16)bReg[c][3];
            *(v4bf*)&Bs[n * LDSP + kk] = p;
        }
    };

    v8f acc[2][4] = {};

    loadTile(0);
    for (int k0 = 0; k0 < K; k0 += KT) {
        __syncthreads();                 // LDS free (prev compute done)
        storeTile();
        __syncthreads();                 // tile visible
        if (k0 + KT < K) loadTile(k0 + KT);   // async: hides under WMMAs below

#pragma unroll
        for (int ks = 0; ks < KT; ks += 32) {
            // B frags: lane -> col n=l16; lanes 0-15: K ks..ks+15, lanes 16-31: +16
            v16bf bfrag[4];
#pragma unroll
            for (int ni = 0; ni < 4; ++ni) {
                int nloc = wave_n * 64 + ni * 16 + l16;
                const bf16* pb = &Bs[nloc * LDSP + ks + half * 16];
                union { v16bf v; v8bf h[2]; } ub;
                ub.h[0] = *(const v8bf*)(pb);
                ub.h[1] = *(const v8bf*)(pb + 8);
                bfrag[ni] = ub.v;
            }
            // A frags: lane -> row M=l16; lanes 0-15: K ks..ks+15, lanes 16-31: +16
#pragma unroll
            for (int mi = 0; mi < 2; ++mi) {
                int rloc = wave_m * 32 + mi * 16 + l16;
                const bf16* pa = &As[rloc * LDSP + ks + half * 16];
                union { v16bf v; v8bf h[2]; } ua;
                ua.h[0] = *(const v8bf*)(pa);
                ua.h[1] = *(const v8bf*)(pa + 8);
#pragma unroll
                for (int ni = 0; ni < 4; ++ni) {
                    acc[mi][ni] = __builtin_amdgcn_wmma_f32_16x16x32_bf16(
                        false, ua.v, false, bfrag[ni], (short)0, acc[mi][ni],
                        false, false);
                }
            }
        }
    }

    // ---- epilogue: C element j <-> M = j + 8*(lane>=16), N = lane%16 ----
#pragma unroll
    for (int ni = 0; ni < 4; ++ni) {
        int col = base_n + wave_n * 64 + ni * 16 + l16;
        float bv = bias ? bias[col] : 0.0f;
#pragma unroll
        for (int mi = 0; mi < 2; ++mi) {
            int m0 = base_m + wave_m * 32 + mi * 16 + half * 8;
            v8f a = acc[mi][ni];
#pragma unroll
            for (int j = 0; j < 8; ++j) {
                float v = a[j] + bv;
                if (relu) v = fmaxf(v, 0.0f);
                C[(size_t)(m0 + j) * N + col] = v;
            }
        }
    }
}

// =====================================================================
// Embedding: x[b,t,:] = tok_emb[idx[b,t],:] + pos_emb[t,:]
// =====================================================================
__global__ __launch_bounds__(256)
void embed_kernel(const int* __restrict__ idx, const float* __restrict__ tok,
                  const float* __restrict__ pos, float* __restrict__ X) {
    int row = blockIdx.x;            // 0..BT-1
    int t   = row % Tc;
    const float* te = tok + (size_t)idx[row] * Ec;
    const float* pe = pos + (size_t)t * Ec;
    float* x = X + (size_t)row * Ec;
    for (int e = threadIdx.x; e < Ec; e += 256) x[e] = te[e] + pe[e];
}

// =====================================================================
// LayerNorm over last dim (E=768), one block per row
// =====================================================================
__global__ __launch_bounds__(256)
void ln_kernel(const float* __restrict__ X, const float* __restrict__ g,
               const float* __restrict__ b, float* __restrict__ Y) {
    __shared__ float red[256];
    int row = blockIdx.x, tid = threadIdx.x;
    const float* x = X + (size_t)row * Ec;

    float s = 0.f;
    for (int e = tid; e < Ec; e += 256) s += x[e];
    red[tid] = s; __syncthreads();
    for (int o = 128; o; o >>= 1) { if (tid < o) red[tid] += red[tid + o]; __syncthreads(); }
    float mu = red[0] / (float)Ec;
    __syncthreads();

    float v = 0.f;
    for (int e = tid; e < Ec; e += 256) { float d = x[e] - mu; v += d * d; }
    red[tid] = v; __syncthreads();
    for (int o = 128; o; o >>= 1) { if (tid < o) red[tid] += red[tid + o]; __syncthreads(); }
    float rstd = rsqrtf(red[0] / (float)Ec + 1e-5f);
    __syncthreads();

    float* y = Y + (size_t)row * Ec;
    for (int e = tid; e < Ec; e += 256) y[e] = (x[e] - mu) * rstd * g[e] + b[e];
}

// =====================================================================
// Streaming causal attention with online softmax.
// One wave32 per query row; lane owns head dims {lane, lane+32}.
// Avoids materializing the 100MB score tensor.
// =====================================================================
__global__ __launch_bounds__(256)
void attn_kernel(const float* __restrict__ Q, const float* __restrict__ Kb,
                 const float* __restrict__ Vb, float* __restrict__ O,
                 float scale) {
    int lane = threadIdx.x & 31;
    int wid  = threadIdx.x >> 5;
    int gidx = blockIdx.x * 8 + wid;        // over B*H*T
    int qt   = gidx % Tc;
    int bh   = gidx / Tc;
    int h    = bh % Hc;
    int b    = bh / Hc;

    const float* qrow = Q + ((size_t)(b * Tc + qt)) * Ec + h * HSc;
    float q0 = qrow[lane], q1 = qrow[lane + 32];

    float m = -__builtin_inff(), lsum = 0.f, acc0 = 0.f, acc1 = 0.f;
    for (int k = 0; k <= qt; ++k) {
        const float* krow = Kb + ((size_t)(b * Tc + k)) * Ec + h * HSc;
        float d = q0 * krow[lane] + q1 * krow[lane + 32];
        for (int off = 16; off; off >>= 1) d += __shfl_xor(d, off, 32);
        float s  = d * scale;
        float mn = fmaxf(m, s);
        float c  = __expf(m - mn);          // first iter: exp(-inf)=0
        float p  = __expf(s - mn);
        const float* vrow = Vb + ((size_t)(b * Tc + k)) * Ec + h * HSc;
        lsum = lsum * c + p;
        acc0 = acc0 * c + p * vrow[lane];
        acc1 = acc1 * c + p * vrow[lane + 32];
        m = mn;
    }
    float inv = 1.f / lsum;
    float* orow = O + ((size_t)(b * Tc + qt)) * Ec + h * HSc;
    orow[lane]      = acc0 * inv;
    orow[lane + 32] = acc1 * inv;
}

// =====================================================================
// Residual add
// =====================================================================
__global__ __launch_bounds__(256)
void add_kernel(float* __restrict__ X, const float* __restrict__ Y, int n) {
    int i = blockIdx.x * 256 + threadIdx.x;
    if (i < n) X[i] += Y[i];
}

// =====================================================================
// Per-row NLL: lse - logit[target]; then mean over rows
// =====================================================================
__global__ __launch_bounds__(256)
void rowloss_kernel(const float* __restrict__ logits, const int* __restrict__ targets,
                    float* __restrict__ partial) {
    __shared__ float red[256];
    int row = blockIdx.x, tid = threadIdx.x;
    const float* lg = logits + (size_t)row * Vc;

    float mx = -__builtin_inff();
    for (int v = tid; v < Vc; v += 256) mx = fmaxf(mx, lg[v]);
    red[tid] = mx; __syncthreads();
    for (int o = 128; o; o >>= 1) { if (tid < o) red[tid] = fmaxf(red[tid], red[tid + o]); __syncthreads(); }
    mx = red[0]; __syncthreads();

    float s = 0.f;
    for (int v = tid; v < Vc; v += 256) s += __expf(lg[v] - mx);
    red[tid] = s; __syncthreads();
    for (int o = 128; o; o >>= 1) { if (tid < o) red[tid] += red[tid + o]; __syncthreads(); }

    if (tid == 0) {
        float lse = mx + __logf(red[0]);
        partial[row] = lse - lg[targets[row]];
    }
}

__global__ __launch_bounds__(256)
void meanloss_kernel(const float* __restrict__ partial, float* __restrict__ out) {
    __shared__ float red[256];
    int tid = threadIdx.x;
    float s = 0.f;
    for (int i = tid; i < BTc; i += 256) s += partial[i];
    red[tid] = s; __syncthreads();
    for (int o = 128; o; o >>= 1) { if (tid < o) red[tid] += red[tid + o]; __syncthreads(); }
    if (tid == 0) out[0] = red[0] / (float)BTc;
}

// =====================================================================
// Orchestration
// =====================================================================
extern "C" void kernel_launch(void* const* d_in, const int* in_sizes, int n_in,
                              void* d_out, int out_size, void* d_ws, size_t ws_size,
                              hipStream_t stream) {
    const int*   idx     = (const int*)  d_in[0];
    const int*   targets = (const int*)  d_in[1];
    const float* tok_emb = (const float*)d_in[2];
    const float* pos_emb = (const float*)d_in[3];
    const float* ln1_g   = (const float*)d_in[4];
    const float* ln1_b   = (const float*)d_in[5];
    const float* wq      = (const float*)d_in[6];
    const float* wk      = (const float*)d_in[7];
    const float* wv      = (const float*)d_in[8];
    const float* wproj   = (const float*)d_in[9];
    const float* bproj   = (const float*)d_in[10];
    const float* ln2_g   = (const float*)d_in[11];
    const float* ln2_b   = (const float*)d_in[12];
    const float* w1      = (const float*)d_in[13];
    const float* b1      = (const float*)d_in[14];
    const float* w2      = (const float*)d_in[15];
    const float* b2      = (const float*)d_in[16];
    const float* lnf_g   = (const float*)d_in[17];
    const float* lnf_b   = (const float*)d_in[18];
    const float* lm_w    = (const float*)d_in[19];
    const float* lm_b    = (const float*)d_in[20];

    float* logits = (float*)d_out;                       // [BT, V]
    float* loss   = logits + (size_t)BTc * Vc;           // scalar

    // workspace layout (floats): ~69 MB
    float* ws  = (float*)d_ws;
    float* X   = ws;                        // BT*E   residual stream
    float* Hh  = X  + (size_t)BTc * Ec;     // BT*E   LN output
    float* Qb  = Hh + (size_t)BTc * Ec;     // BT*E
    float* Kb  = Qb + (size_t)BTc * Ec;     // BT*E
    float* Vb  = Kb + (size_t)BTc * Ec;     // BT*E
    float* A1  = Vb + (size_t)BTc * Ec;     // BT*E
    float* A2  = A1 + (size_t)BTc * Ec;     // BT*E
    float* Mh  = A2 + (size_t)BTc * Ec;     // BT*4E  MLP hidden
    float* rwl = Mh + (size_t)BTc * 4 * Ec; // BT     per-row loss

    const float scale = 1.0f / sqrtf((float)Ec);   // reference scales by E^-0.5

    embed_kernel<<<BTc, 256, 0, stream>>>(idx, tok_emb, pos_emb, X);

    dim3 gEE(Ec / 128, BTc / 128);          // N=768 tiles
    dim3 gE4(4 * Ec / 128, BTc / 128);      // N=3072 tiles
    dim3 gV (Vc / 128, BTc / 128);          // N=32000 tiles

    for (int l = 0; l < Lc; ++l) {
        ln_kernel<<<BTc, 256, 0, stream>>>(X, ln1_g + l * Ec, ln1_b + l * Ec, Hh);

        gemm_bf16_wmma<<<gEE, 256, 0, stream>>>(Hh, wq + (size_t)l * Ec * Ec, nullptr, Qb, BTc, Ec, Ec, 0);
        gemm_bf16_wmma<<<gEE, 256, 0, stream>>>(Hh, wk + (size_t)l * Ec * Ec, nullptr, Kb, BTc, Ec, Ec, 0);
        gemm_bf16_wmma<<<gEE, 256, 0, stream>>>(Hh, wv + (size_t)l * Ec * Ec, nullptr, Vb, BTc, Ec, Ec, 0);

        attn_kernel<<<(Bc * Hc * Tc) / 8, 256, 0, stream>>>(Qb, Kb, Vb, A1, scale);

        // projection applied twice (faithful to reference)
        gemm_bf16_wmma<<<gEE, 256, 0, stream>>>(A1, wproj + (size_t)l * Ec * Ec, bproj + l * Ec, A2, BTc, Ec, Ec, 0);
        gemm_bf16_wmma<<<gEE, 256, 0, stream>>>(A2, wproj + (size_t)l * Ec * Ec, bproj + l * Ec, A1, BTc, Ec, Ec, 0);
        add_kernel<<<(BTc * Ec) / 256, 256, 0, stream>>>(X, A1, BTc * Ec);

        ln_kernel<<<BTc, 256, 0, stream>>>(X, ln2_g + l * Ec, ln2_b + l * Ec, Hh);
        gemm_bf16_wmma<<<gE4, 256, 0, stream>>>(Hh, w1 + (size_t)l * Ec * 4 * Ec, b1 + l * 4 * Ec, Mh, BTc, 4 * Ec, Ec, 1);
        gemm_bf16_wmma<<<gEE, 256, 0, stream>>>(Mh, w2 + (size_t)l * 4 * Ec * Ec, b2 + l * Ec, A2, BTc, Ec, 4 * Ec, 0);
        add_kernel<<<(BTc * Ec) / 256, 256, 0, stream>>>(X, A2, BTc * Ec);
    }

    ln_kernel<<<BTc, 256, 0, stream>>>(X, lnf_g, lnf_b, Hh);
    gemm_bf16_wmma<<<gV, 256, 0, stream>>>(Hh, lm_w, lm_b, logits, BTc, Vc, Ec, 0);

    rowloss_kernel<<<BTc, 256, 0, stream>>>(logits, targets, rwl);
    meanloss_kernel<<<1, 256, 0, stream>>>(rwl, loss);
}
